// PeriodicPrimitives3D_76751065579746
// MI455X (gfx1250) — matrix operand order repaired and checked by reading, not compiled, not run on hardware
//
#include <hip/hip_runtime.h>
#include <hip/hip_bf16.h>
#include <math.h>

typedef __attribute__((ext_vector_type(16))) _Float16 v16h;
typedef __attribute__((ext_vector_type(8)))  float    v8f;

// ---------------- constants (match reference) ----------------
#define NP        512                 // padded primitive count (power of 2)
#define IMGW      200
#define IMGH      200
#define HWPIX     (IMGW * IMGH)       // 40000
#define TOPK      8
#define NEARP     0.1f
#define COVDIL    0.3f
#define TANHALF   0.57735026918962576f          // tan(30 deg)
#define FXC       (IMGW / (2.0f * TANHALF))
#define FYC       (IMGH / (2.0f * TANHALF))
// omega = 2*pi * (idx * 128/1024) / W   (folds the /W of tcoord into the phase)
#define OMEGA_SCALE (0.125f * 6.2831853071795864f / (float)IMGW)

// workspace layout (floats):
//   [0        .. 24*NP)   : unsorted SoA scratch  (8 params, 8 omega, 8 coeff)
//   [24*NP    .. 48*NP)   : depth-sorted SoA      (same layout)
//   [48*NP    .. +768)    : sorted colors as _Float16 [NP][3]
#define WS_SORTED (24 * NP)
#define WS_COLF16 (48 * NP)

// ---------------------------------------------------------------------------
// Kernel 1: per-primitive preprocessing + in-LDS bitonic depth argsort.
// Single block of NP threads.
// ---------------------------------------------------------------------------
__global__ __launch_bounds__(NP) void prep_sort_kernel(
    const float* __restrict__ colors, const float* __restrict__ opac,
    const float* __restrict__ pos,    const float* __restrict__ scl,
    const float* __restrict__ rot,    const float* __restrict__ coef,
    const int*   __restrict__ widx,   const float* __restrict__ view,
    const float* __restrict__ VP,     float* __restrict__ ws, int N)
{
    const int t = threadIdx.x;
    __shared__ float skey[NP];
    __shared__ int   sval[NP];

    float* U = ws;                 // unsorted
    float* S = ws + WS_SORTED;     // sorted
    _Float16* scolS = (_Float16*)(ws + WS_COLF16);

    float depth = 1e30f;
    if (t < N) {
        // quaternion -> rotation matrix
        float q0 = rot[t*4+0], q1 = rot[t*4+1], q2 = rot[t*4+2], q3 = rot[t*4+3];
        float qn = sqrtf(q0*q0 + q1*q1 + q2*q2 + q3*q3) + 1e-8f;
        float r = q0/qn, x = q1/qn, y = q2/qn, z = q3/qn;
        float R[3][3] = {
            {1.0f - 2.0f*(y*y + z*z), 2.0f*(x*y - r*z),        2.0f*(x*z + r*y)},
            {2.0f*(x*y + r*z),        1.0f - 2.0f*(x*x + z*z), 2.0f*(y*z - r*x)},
            {2.0f*(x*z - r*y),        2.0f*(y*z + r*x),        1.0f - 2.0f*(x*x + y*y)}};
        // Sigma = R diag(s^2) R^T
        float s2[3];
        #pragma unroll
        for (int j = 0; j < 3; ++j) { float e = __expf(scl[t*3+j]); s2[j] = e*e; }
        float Sg[3][3];
        #pragma unroll
        for (int i = 0; i < 3; ++i)
            #pragma unroll
            for (int k = 0; k < 3; ++k)
                Sg[i][k] = R[i][0]*s2[0]*R[k][0] + R[i][1]*s2[1]*R[k][1] + R[i][2]*s2[2]*R[k][2];

        float ph[4] = {pos[t*3+0], pos[t*3+1], pos[t*3+2], 1.0f};
        float tc[3], clip[4];
        #pragma unroll
        for (int i = 0; i < 3; ++i)
            tc[i] = view[i*4+0]*ph[0] + view[i*4+1]*ph[1] + view[i*4+2]*ph[2] + view[i*4+3];
        #pragma unroll
        for (int i = 0; i < 4; ++i)
            clip[i] = VP[i*4+0]*ph[0] + VP[i*4+1]*ph[1] + VP[i*4+2]*ph[2] + VP[i*4+3];

        float wclip = clip[3];
        float wd = (fabsf(wclip) > 1e-6f) ? wclip : 1e-6f;
        float pxv = (clip[0]/wd * 0.5f + 0.5f) * (float)IMGW;
        float pyv = (clip[1]/wd * 0.5f + 0.5f) * (float)IMGH;

        depth = tc[2];
        float tz = fmaxf(depth, NEARP);
        float J[2][3] = {{FXC/tz, 0.0f, -FXC*tc[0]/(tz*tz)},
                         {0.0f, FYC/tz, -FYC*tc[1]/(tz*tz)}};
        float M[2][3];
        #pragma unroll
        for (int a = 0; a < 2; ++a)
            #pragma unroll
            for (int c = 0; c < 3; ++c)
                M[a][c] = J[a][0]*view[0*4+c] + J[a][1]*view[1*4+c] + J[a][2]*view[2*4+c];
        float tmp[2][3];
        #pragma unroll
        for (int a = 0; a < 2; ++a)
            #pragma unroll
            for (int c = 0; c < 3; ++c)
                tmp[a][c] = M[a][0]*Sg[0][c] + M[a][1]*Sg[1][c] + M[a][2]*Sg[2][c];
        float cov00 = tmp[0][0]*M[0][0] + tmp[0][1]*M[0][1] + tmp[0][2]*M[0][2] + COVDIL;
        float cov01 = tmp[0][0]*M[1][0] + tmp[0][1]*M[1][1] + tmp[0][2]*M[1][2];
        float cov11 = tmp[1][0]*M[1][0] + tmp[1][1]*M[1][1] + tmp[1][2]*M[1][2] + COVDIL;
        float det = fmaxf(cov00*cov11 - cov01*cov01, 1e-12f);
        float cA =  cov11/det, cB = -cov01/det, cC = cov00/det;

        // anisotropy axis: a2 = normalize(M @ R[:,0])
        float a3x = R[0][0], a3y = R[1][0], a3z = R[2][0];
        float a2x = M[0][0]*a3x + M[0][1]*a3y + M[0][2]*a3z;
        float a2y = M[1][0]*a3x + M[1][1]*a3y + M[1][2]*a3z;
        float an = sqrtf(a2x*a2x + a2y*a2y) + 1e-8f;
        a2x /= an; a2y /= an;

        bool valid = (depth > NEARP) && (wclip > 1e-4f);
        float op = valid ? opac[t] : 0.0f;

        U[0*NP+t] = pxv; U[1*NP+t] = pyv;
        U[2*NP+t] = cA;  U[3*NP+t] = cB;  U[4*NP+t] = cC;
        U[5*NP+t] = a2x; U[6*NP+t] = a2y; U[7*NP+t] = op;
        #pragma unroll
        for (int k = 0; k < TOPK; ++k) {
            U[ 8*NP + t*TOPK + k] = (float)widx[t*TOPK+k] * OMEGA_SCALE;
            U[16*NP + t*TOPK + k] = coef[t*TOPK+k];
        }
    }
    skey[t] = depth;
    sval[t] = t;
    __threadfence();
    __syncthreads();

    // bitonic argsort (ascending by depth)
    for (int k = 2; k <= NP; k <<= 1) {
        for (int j = k >> 1; j > 0; j >>= 1) {
            int ixj = t ^ j;
            if (ixj > t) {
                bool up = ((t & k) == 0);
                float ka = skey[t], kb = skey[ixj];
                if ((ka > kb) == up) {
                    skey[t] = kb; skey[ixj] = ka;
                    int tv = sval[t]; sval[t] = sval[ixj]; sval[ixj] = tv;
                }
            }
            __syncthreads();
        }
    }

    // gather-write sorted SoA
    int o = sval[t];
    if (o < N) {
        #pragma unroll
        for (int a = 0; a < 8; ++a) S[a*NP+t] = U[a*NP+o];
        #pragma unroll
        for (int k = 0; k < TOPK; ++k) {
            S[ 8*NP + t*TOPK + k] = U[ 8*NP + o*TOPK + k];
            S[16*NP + t*TOPK + k] = U[16*NP + o*TOPK + k];
        }
        #pragma unroll
        for (int c = 0; c < 3; ++c) scolS[t*3+c] = (_Float16)colors[o*3+c];
    } else {
        #pragma unroll
        for (int a = 0; a < 8; ++a) S[a*NP+t] = 0.0f;
        #pragma unroll
        for (int k = 0; k < TOPK; ++k) { S[8*NP+t*TOPK+k] = 0.0f; S[16*NP+t*TOPK+k] = 0.0f; }
        #pragma unroll
        for (int c = 0; c < 3; ++c) scolS[t*3+c] = (_Float16)0.0f;
    }
}

// ---------------------------------------------------------------------------
// Kernel 2: fused per-pixel compositing.
// 256 threads = 8 waves; each wave owns 32 pixels. Weights are generated with
// a sequential transmittance scan per lane, staged into LDS as f16, then the
// colors x weights contraction runs on v_wmma_f32_16x16x32_f16 (2 tiles/wave
// per 32-primitive chunk).
// ---------------------------------------------------------------------------
#define WROW 40   // padded LDS row stride (halfs) to avoid bank conflicts

__global__ __launch_bounds__(256) void render_kernel(
    const float* __restrict__ ws, const float* __restrict__ bg,
    float* __restrict__ out)
{
    const float* S    = ws + WS_SORTED;
    const float* gOm  = S + 8*NP;      // [NP][8] omega (sorted)
    const float* gCf  = S + 16*NP;     // [NP][8] coeff (sorted)

    __shared__ float spx[NP], spy[NP], scA[NP], scB[NP], scC[NP];
    __shared__ float sax[NP], say[NP], sop[NP];
    __shared__ _Float16 scol[NP*3];
    __shared__ _Float16 wbuf[8 * 32 * WROW];
    __shared__ float stf[256];

    const int tid = threadIdx.x;
    const _Float16* gCol = (const _Float16*)(ws + WS_COLF16);
    for (int i = tid; i < NP; i += 256) {
        spx[i] = S[0*NP+i]; spy[i] = S[1*NP+i];
        scA[i] = S[2*NP+i]; scB[i] = S[3*NP+i]; scC[i] = S[4*NP+i];
        sax[i] = S[5*NP+i]; say[i] = S[6*NP+i]; sop[i] = S[7*NP+i];
        scol[i*3+0] = gCol[i*3+0];
        scol[i*3+1] = gCol[i*3+1];
        scol[i*3+2] = gCol[i*3+2];
    }
    __syncthreads();

    const int wave = tid >> 5;
    const int lane = tid & 31;
    const int chan = lane & 15;              // D-matrix column = color channel
    const int cc   = (chan < 3) ? chan : 2;  // clamped (safe speculative loads)
    const int k0   = (lane < 16) ? 0 : 8;    // A-matrix K base (f16 16x32 layout)
    const int kb   = (lane < 16) ? 0 : 16;   // unused: B uses k0, same k order as A

    const int pix = blockIdx.x * 256 + tid;  // this lane's pixel
    const float gx = (float)(pix % IMGW) + 0.5f;
    const float gy = (float)(pix / IMGW) + 0.5f;

    float T = 1.0f;
    v8f acc0 = {}; v8f acc1 = {};

    _Float16* wrow = &wbuf[(wave * 32 + lane) * WROW];

    for (int c = 0; c < NP / 32; ++c) {
        // ---- phase A: weights for 32 primitives (sequential T scan) ----
        for (int k = 0; k < 32; ++k) {
            const int i = c * 32 + k;
            const float dx = spx[i] - gx;
            const float dy = spy[i] - gy;
            float pw = -0.5f * (scA[i]*dx*dx + scC[i]*dy*dy) - scB[i]*dx*dy;
            const float G = __expf(fminf(pw, 0.0f));
            const float tcv = dx * sax[i] + dy * say[i];   // /W folded into omega
            float wv = 0.0f;
            #pragma unroll
            for (int k2 = 0; k2 < TOPK; ++k2)
                wv += gCf[i*TOPK+k2] * __cosf(gOm[i*TOPK+k2] * tcv);
            const float alpha = fminf(fmaxf(sop[i] * G * wv, 0.0f), 0.99f);
            wrow[k] = (_Float16)(alpha * T);
            T *= (1.0f - alpha);
        }
        __syncthreads();

        // ---- phase B: D += W(16x32) x C(32x16) on the matrix pipe ----
        v16h b;
        #pragma unroll
        for (int j = 0; j < 16; ++j) {
            const int kk = c * 32 + k0 + (j & 7) + ((j >> 3) << 4);
            const _Float16 cv = scol[kk*3 + cc];
            b[j] = (chan < 3) ? cv : (_Float16)0.0f;
        }
        #pragma unroll
        for (int tt = 0; tt < 2; ++tt) {
            const int row = tt * 16 + (lane & 15);
            const _Float16* ar = &wbuf[(wave * 32 + row) * WROW + k0];
            v16h a;
            #pragma unroll
            for (int j = 0; j < 8; ++j) { a[j] = ar[j]; a[8+j] = ar[16+j]; }
            if (tt == 0)
                acc0 = __builtin_amdgcn_wmma_f32_16x16x32_f16(
                        false, a, false, b, (short)0, acc0, false, false);
            else
                acc1 = __builtin_amdgcn_wmma_f32_16x16x32_f16(
                        false, a, false, b, (short)0, acc1, false, false);
        }
        __syncthreads();
    }

    stf[tid] = T;            // final transmittance of this lane's pixel
    __syncthreads();

    const float bgc = (chan < 3) ? bg[cc] : 0.0f;
    #pragma unroll
    for (int tt = 0; tt < 2; ++tt) {
        #pragma unroll
        for (int v = 0; v < 8; ++v) {
            const int row = v + ((lane < 16) ? 0 : 8);
            const int piw = tt * 16 + row;                 // pixel within wave
            const int gp  = blockIdx.x * 256 + wave * 32 + piw;
            const float dv = (tt == 0) ? acc0[v] : acc1[v];
            if (chan < 3 && gp < HWPIX)
                out[chan * HWPIX + gp] = dv + bgc * stf[wave * 32 + piw];
        }
    }
}

// ---------------------------------------------------------------------------
extern "C" void kernel_launch(void* const* d_in, const int* in_sizes, int n_in,
                              void* d_out, int out_size, void* d_ws, size_t ws_size,
                              hipStream_t stream) {
    const float* colors = (const float*)d_in[0];
    const float* opac   = (const float*)d_in[1];
    const float* bg     = (const float*)d_in[2];
    const float* pos    = (const float*)d_in[3];
    const float* scl    = (const float*)d_in[4];
    const float* rot    = (const float*)d_in[5];
    const float* coef   = (const float*)d_in[6];
    const int*   widx   = (const int*)d_in[7];
    // d_in[8] = cam_position (unused)
    const float* view   = (const float*)d_in[9];
    const float* VP     = (const float*)d_in[10];
    // d_in[11]/[12] = image_width/height (compile-time constants 200x200)

    int N = in_sizes[3] / 3;
    if (N > NP) N = NP;

    float* ws = (float*)d_ws;
    prep_sort_kernel<<<1, NP, 0, stream>>>(colors, opac, pos, scl, rot, coef,
                                           widx, view, VP, ws, N);
    render_kernel<<<(HWPIX + 255) / 256, 256, 0, stream>>>(ws, bg, (float*)d_out);
}
